// VideoTransformer_37838661877850
// MI455X (gfx1250) — compile-verified, matrix operand order checked by reading, not run on hardware
//
#include <hip/hip_runtime.h>
#include <hip/hip_bf16.h>
#include <math.h>

typedef __attribute__((ext_vector_type(16))) _Float16 v16h;
typedef __attribute__((ext_vector_type(8)))  _Float16 v8h;
typedef __attribute__((ext_vector_type(8)))  float    v8f;

#define D_MODEL 512
#define SEQ     512
#define BATCH   32
#define NHEAD   8
#define DFF     2048
#define FEAT    2048
#define MROWS   (BATCH * SEQ)   // 16384

union F16x16 {
    v16h v;
    v8h  h2[2];
    _Float16 e[16];
};

// Element index e of a 16-bit A/B fragment -> K index (16-bit A-matrix 16x32 layout,
// lanes 0-15: K groups {0..7,16..23}, lanes 16-31: K groups {8..15,24..31})
__device__ __forceinline__ int fragK(int e, int half) {
    return (e < 8) ? (half * 8 + e) : (16 + half * 8 + (e - 8));
}

// Load a 16x32 f16 fragment from an LDS row (row stride keeps 16B alignment).
__device__ __forceinline__ v16h load_frag(const _Float16* row, int half) {
    F16x16 u;
    u.h2[0] = *(const v8h*)(row + half * 8);
    u.h2[1] = *(const v8h*)(row + 16 + half * 8);
    return u.v;
}

__device__ __forceinline__ v8f wmma_f16(v16h a, v16h b, v8f c) {
    return __builtin_amdgcn_wmma_f32_16x16x32_f16(false, a, false, b, (short)0, c,
                                                  false, false);
}

// ---------------------------------------------------------------------------
// C[M,N] = epilogue(A[M,K] @ W[K,N] + bias[N])
// 128x128 tile / block, 8 waves x (32x64), double-buffered LDS, K-stage 32.
// mode: 0 = bias, 1 = bias+ReLU, 2 = bias + positional encoding
// ---------------------------------------------------------------------------
__device__ __forceinline__ void gemm_load_regs(const float* __restrict__ A,
                                               const float* __restrict__ W,
                                               int t, int bm, int bn, int K, int N,
                                               int k0, F16x16& ra, F16x16& rb) {
    const int r   = t & 127;          // A-row / B-col handled by this thread
    const int kk0 = (t >> 7) * 16;    // 16-wide K slice
    const float* sa = A + (size_t)(bm + r) * K + k0 + kk0;
    #pragma unroll
    for (int j = 0; j < 16; ++j) ra.e[j] = (_Float16)sa[j];          // 64B contig
    const float* sb = W + (size_t)(k0 + kk0) * N + bn + r;
    #pragma unroll
    for (int j = 0; j < 16; ++j) rb.e[j] = (_Float16)sb[(size_t)j * N]; // coalesced in n
}

__device__ __forceinline__ void gemm_store_lds(_Float16 (*As)[40], _Float16 (*Bs)[40],
                                               int t, const F16x16& ra, const F16x16& rb) {
    const int r   = t & 127;
    const int kk0 = (t >> 7) * 16;
    *(v8h*)&As[r][kk0]     = ra.h2[0];
    *(v8h*)&As[r][kk0 + 8] = ra.h2[1];
    *(v8h*)&Bs[r][kk0]     = rb.h2[0];   // Bs stored transposed: [n][k]
    *(v8h*)&Bs[r][kk0 + 8] = rb.h2[1];
}

__global__ __launch_bounds__(256) void gemm_kernel(const float* __restrict__ A,
                                                   const float* __restrict__ W,
                                                   const float* __restrict__ bias,
                                                   float* __restrict__ C,
                                                   int M, int N, int K, int mode) {
    __shared__ _Float16 As[2][128][40];   // stride 40 halves = 80B (16B aligned)
    __shared__ _Float16 Bs[2][128][40];

    const int t    = threadIdx.x;
    const int bm   = blockIdx.y * 128;
    const int bn   = blockIdx.x * 128;
    const int w    = t >> 5;
    const int lane = t & 31;
    const int wy   = w >> 1;          // 0..3 : 32-row band
    const int wx   = w & 1;           // 0..1 : 64-col band
    const int m16  = lane & 15;
    const int half = lane >> 4;

    v8f acc[2][4] = {};

    // Pipeline prologue: stage 0 -> LDS buffer 0
    F16x16 ra, rb;
    gemm_load_regs(A, W, t, bm, bn, K, N, 0, ra, rb);
    gemm_store_lds(As[0], Bs[0], t, ra, rb);
    __syncthreads();

    const int nstages = K >> 5;
    for (int s = 0; s < nstages; ++s) {
        const int cur = s & 1;

        if (s + 2 < nstages) {        // L2 prefetch two stages ahead
            const int r = t & 127, kk0 = (t >> 7) * 16;
            __builtin_prefetch(A + (size_t)(bm + r) * K + (s + 2) * 32 + kk0, 0, 1);
            __builtin_prefetch(W + (size_t)((s + 2) * 32 + kk0) * N + bn + r, 0, 1);
        }
        if (s + 1 < nstages)          // issue next stage's global loads early
            gemm_load_regs(A, W, t, bm, bn, K, N, (s + 1) * 32, ra, rb);

        // 8 WMMAs against the current buffer
        v16h a0 = load_frag(&As[cur][wy * 32 + m16][0], half);
        v16h a1 = load_frag(&As[cur][wy * 32 + 16 + m16][0], half);
        #pragma unroll
        for (int nt = 0; nt < 4; ++nt) {
            v16h b = load_frag(&Bs[cur][wx * 64 + nt * 16 + m16][0], half);
            acc[0][nt] = wmma_f16(a0, b, acc[0][nt]);
            acc[1][nt] = wmma_f16(a1, b, acc[1][nt]);
        }

        if (s + 1 < nstages)          // drain staged regs into the other buffer
            gemm_store_lds(As[cur ^ 1], Bs[cur ^ 1], t, ra, rb);
        __syncthreads();
    }

    #pragma unroll
    for (int mt = 0; mt < 2; ++mt) {
        #pragma unroll
        for (int v = 0; v < 8; ++v) {
            const int row  = bm + wy * 32 + mt * 16 + v + 8 * half;
            const int srow = row & (SEQ - 1);
            #pragma unroll
            for (int nt = 0; nt < 4; ++nt) {
                const int col = bn + wx * 64 + nt * 16 + m16;
                float x = acc[mt][nt][v] + bias[col];
                if (mode == 1) {
                    x = fmaxf(x, 0.0f);
                } else if (mode == 2) {
                    int i = col & ~1;
                    float dv  = __expf(-(float)i * (9.210340371976184f / (float)D_MODEL));
                    float ang = (float)srow * dv;
                    x += (col & 1) ? __cosf(ang) : __sinf(ang);
                }
                C[(size_t)row * N + col] = x;
            }
        }
    }
}

// ---------------------------------------------------------------------------
// Flash attention: grid (B*H, S/64), block 128 (4 waves); wave = 16 query rows.
// ---------------------------------------------------------------------------
__global__ __launch_bounds__(128) void attn_kernel(const float* __restrict__ Q,
                                                   const float* __restrict__ K,
                                                   const float* __restrict__ V,
                                                   float* __restrict__ O) {
    __shared__ _Float16 P[4][16][40];  // per-wave P tile (C-layout -> A-layout relay)

    const int D = D_MODEL;
    const int w    = threadIdx.x >> 5;
    const int lane = threadIdx.x & 31;
    const int m16  = lane & 15;
    const int half = lane >> 4;

    const int bh   = blockIdx.x;
    const int b    = bh >> 3;
    const int hh   = bh & 7;
    const int hoff = hh * 64;
    const int qbase = (blockIdx.y * 4 + w) * 16;

    const float* Qb = Q + (size_t)b * SEQ * D;
    const float* Kb = K + (size_t)b * SEQ * D;
    const float* Vb = V + (size_t)b * SEQ * D;
    float*       Ob = O + (size_t)b * SEQ * D;

    // Q fragments (d_k = 64 -> two 32-wide k-steps), scale 1/sqrt(64) folded in.
    F16x16 aq[2];
    #pragma unroll
    for (int j = 0; j < 2; ++j)
        #pragma unroll
        for (int e = 0; e < 16; ++e) {
            int d = j * 32 + fragK(e, half);
            aq[j].e[e] = (_Float16)(Qb[(size_t)(qbase + m16) * D + hoff + d] * 0.125f);
        }

    v8f acc[4] = {};                 // ctx, 4 x 16 value-dim columns
    float mrow[8], lrow[8];
    #pragma unroll
    for (int v = 0; v < 8; ++v) { mrow[v] = -INFINITY; lrow[v] = 0.0f; }

    for (int kc = 0; kc < SEQ; kc += 32) {
        v8f s0 = {};
        v8f s1 = {};
        #pragma unroll
        for (int j = 0; j < 2; ++j) {
            F16x16 bk0, bk1;
            #pragma unroll
            for (int e = 0; e < 16; ++e) {
                int d = j * 32 + fragK(e, half);
                bk0.e[e] = (_Float16)Kb[(size_t)(kc + m16) * D + hoff + d];
                bk1.e[e] = (_Float16)Kb[(size_t)(kc + 16 + m16) * D + hoff + d];
            }
            s0 = wmma_f16(aq[j].v, bk0.v, s0);
            s1 = wmma_f16(aq[j].v, bk1.v, s1);
        }

        // Online softmax: row r = v + 8*half spans lanes (r%16) -> xor-shuffle reduce.
        #pragma unroll
        for (int v = 0; v < 8; ++v) {
            float mx = fmaxf(s0[v], s1[v]);
            mx = fmaxf(mx, __shfl_xor(mx, 1));
            mx = fmaxf(mx, __shfl_xor(mx, 2));
            mx = fmaxf(mx, __shfl_xor(mx, 4));
            mx = fmaxf(mx, __shfl_xor(mx, 8));
            float mnew  = fmaxf(mrow[v], mx);
            float alpha = __expf(mrow[v] - mnew);
            float p0 = __expf(s0[v] - mnew);
            float p1 = __expf(s1[v] - mnew);
            float ss = p0 + p1;
            ss += __shfl_xor(ss, 1);
            ss += __shfl_xor(ss, 2);
            ss += __shfl_xor(ss, 4);
            ss += __shfl_xor(ss, 8);
            lrow[v] = lrow[v] * alpha + ss;
            mrow[v] = mnew;
            acc[0][v] *= alpha; acc[1][v] *= alpha;
            acc[2][v] *= alpha; acc[3][v] *= alpha;
            P[w][v + 8 * half][m16]      = (_Float16)p0;
            P[w][v + 8 * half][m16 + 16] = (_Float16)p1;
        }
        __syncthreads();

        v16h ap = load_frag(&P[w][m16][0], half);   // P as A-matrix (16x32)
        #pragma unroll
        for (int tt = 0; tt < 4; ++tt) {
            F16x16 bv;
            #pragma unroll
            for (int e = 0; e < 16; ++e) {
                int key = kc + fragK(e, half);
                bv.e[e] = (_Float16)Vb[(size_t)key * D + hoff + tt * 16 + m16];
            }
            acc[tt] = wmma_f16(ap, bv.v, acc[tt]);
        }
        __syncthreads();
    }

    #pragma unroll
    for (int v = 0; v < 8; ++v) {
        float inv = 1.0f / lrow[v];
        int row = qbase + v + 8 * half;
        #pragma unroll
        for (int tt = 0; tt < 4; ++tt)
            Ob[(size_t)row * D + hoff + tt * 16 + m16] = acc[tt][v] * inv;
    }
}

// ---------------------------------------------------------------------------
// y = LayerNorm(x + res) * g + b ; wave per 512-wide row (in-place safe)
// ---------------------------------------------------------------------------
__global__ __launch_bounds__(256) void ln_kernel(const float* __restrict__ x,
                                                 const float* __restrict__ res,
                                                 const float* __restrict__ g,
                                                 const float* __restrict__ bta,
                                                 float* __restrict__ y) {
    const int w    = threadIdx.x >> 5;
    const int lane = threadIdx.x & 31;
    const size_t row = (size_t)blockIdx.x * 8 + w;
    const float* xr = x   + row * D_MODEL;
    const float* rr = res + row * D_MODEL;
    float v[16], s = 0.0f, s2 = 0.0f;
    #pragma unroll
    for (int i = 0; i < 16; ++i) {
        float t = xr[lane + i * 32] + rr[lane + i * 32];
        v[i] = t; s += t; s2 += t * t;
    }
    #pragma unroll
    for (int m = 1; m < 32; m <<= 1) {
        s  += __shfl_xor(s, m);
        s2 += __shfl_xor(s2, m);
    }
    float mu   = s * (1.0f / D_MODEL);
    float var  = s2 * (1.0f / D_MODEL) - mu * mu;
    float rstd = rsqrtf(var + 1e-5f);
    float* yr = y + row * D_MODEL;
    #pragma unroll
    for (int i = 0; i < 16; ++i) {
        int d = lane + i * 32;
        yr[d] = (v[i] - mu) * rstd * g[d] + bta[d];
    }
}

// ---------------------------------------------------------------------------
// out[b] = sigmoid(LN(mean_s h[b,s,:]) @ fcw + fcb) ; one block per batch row
// ---------------------------------------------------------------------------
__global__ __launch_bounds__(256) void head_kernel(const float* __restrict__ h,
                                                   const float* __restrict__ g,
                                                   const float* __restrict__ bta,
                                                   const float* __restrict__ fcw,
                                                   const float* __restrict__ fcb,
                                                   float* __restrict__ out) {
    __shared__ float red[256];
    const int b = blockIdx.x, tid = threadIdx.x;
    const float* hb = h + (size_t)b * SEQ * D_MODEL;
    const int d0 = tid * 2;
    float p0 = 0.0f, p1 = 0.0f;
    for (int s = 0; s < SEQ; ++s) {
        p0 += hb[(size_t)s * D_MODEL + d0];
        p1 += hb[(size_t)s * D_MODEL + d0 + 1];
    }
    p0 *= (1.0f / SEQ); p1 *= (1.0f / SEQ);

    red[tid] = p0 + p1; __syncthreads();
    for (int off = 128; off > 0; off >>= 1) {
        if (tid < off) red[tid] += red[tid + off];
        __syncthreads();
    }
    float mu = red[0] * (1.0f / D_MODEL); __syncthreads();

    red[tid] = p0 * p0 + p1 * p1; __syncthreads();
    for (int off = 128; off > 0; off >>= 1) {
        if (tid < off) red[tid] += red[tid + off];
        __syncthreads();
    }
    float var = red[0] * (1.0f / D_MODEL) - mu * mu; __syncthreads();
    float rstd = rsqrtf(var + 1e-5f);

    float y0 = (p0 - mu) * rstd * g[d0]     + bta[d0];
    float y1 = (p1 - mu) * rstd * g[d0 + 1] + bta[d0 + 1];
    red[tid] = y0 * fcw[d0] + y1 * fcw[d0 + 1]; __syncthreads();
    for (int off = 128; off > 0; off >>= 1) {
        if (tid < off) red[tid] += red[tid + off];
        __syncthreads();
    }
    if (tid == 0) out[b] = 1.0f / (1.0f + __expf(-(red[0] + fcb[0])));
}

// ---------------------------------------------------------------------------
extern "C" void kernel_launch(void* const* d_in, const int* in_sizes, int n_in,
                              void* d_out, int out_size, void* d_ws, size_t ws_size,
                              hipStream_t stream) {
    const float* x    = (const float*)d_in[0];
    const float* Wp   = (const float*)d_in[1];
    const float* bp   = (const float*)d_in[2];
    const float* Wq   = (const float*)d_in[3];
    const float* bq   = (const float*)d_in[4];
    const float* Wk   = (const float*)d_in[5];
    const float* bk   = (const float*)d_in[6];
    const float* Wv   = (const float*)d_in[7];
    const float* bv   = (const float*)d_in[8];
    const float* Wo   = (const float*)d_in[9];
    const float* bo   = (const float*)d_in[10];
    const float* ln1g = (const float*)d_in[11];
    const float* ln1b = (const float*)d_in[12];
    const float* W1   = (const float*)d_in[13];
    const float* b1   = (const float*)d_in[14];
    const float* W2   = (const float*)d_in[15];
    const float* b2   = (const float*)d_in[16];
    const float* ln2g = (const float*)d_in[17];
    const float* ln2b = (const float*)d_in[18];
    const float* lng  = (const float*)d_in[19];
    const float* lnb  = (const float*)d_in[20];
    const float* fcw  = (const float*)d_in[21];
    const float* fcb  = (const float*)d_in[22];
    float* out = (float*)d_out;

    const size_t MD = (size_t)MROWS * D_MODEL;   // 8,388,608 floats
    float* ws  = (float*)d_ws;
    float* h   = ws;
    float* qb  = ws + MD;
    float* kb  = ws + 2 * MD;
    float* vb  = ws + 3 * MD;
    float* cx  = ws + 4 * MD;
    float* ao  = ws + 5 * MD;
    float* ff  = ws + 6 * MD;                    // 4*MD floats (16384 x 2048)

    dim3 blk(256);
    dim3 gD(D_MODEL / 128, MROWS / 128);         // (4, 128)
    dim3 gF(DFF / 128, MROWS / 128);             // (16, 128)
    dim3 gAttn(BATCH * NHEAD, SEQ / 64);         // (256, 8)
    const int lnGrid = MROWS / 8;                // 2048

    // input projection with fused positional encoding (mode 2)
    gemm_kernel<<<gD, blk, 0, stream>>>(x, Wp, bp, h, MROWS, D_MODEL, FEAT, 2);

    for (int l = 0; l < 4; ++l) {
        const float* Wq_l = Wq + (size_t)l * D_MODEL * D_MODEL;
        const float* Wk_l = Wk + (size_t)l * D_MODEL * D_MODEL;
        const float* Wv_l = Wv + (size_t)l * D_MODEL * D_MODEL;
        const float* Wo_l = Wo + (size_t)l * D_MODEL * D_MODEL;
        const float* W1_l = W1 + (size_t)l * D_MODEL * DFF;
        const float* W2_l = W2 + (size_t)l * DFF * D_MODEL;

        gemm_kernel<<<gD, blk, 0, stream>>>(h, Wq_l, bq + l * D_MODEL, qb,
                                            MROWS, D_MODEL, D_MODEL, 0);
        gemm_kernel<<<gD, blk, 0, stream>>>(h, Wk_l, bk + l * D_MODEL, kb,
                                            MROWS, D_MODEL, D_MODEL, 0);
        gemm_kernel<<<gD, blk, 0, stream>>>(h, Wv_l, bv + l * D_MODEL, vb,
                                            MROWS, D_MODEL, D_MODEL, 0);
        attn_kernel<<<gAttn, dim3(128), 0, stream>>>(qb, kb, vb, cx);
        gemm_kernel<<<gD, blk, 0, stream>>>(cx, Wo_l, bo + l * D_MODEL, ao,
                                            MROWS, D_MODEL, D_MODEL, 0);
        ln_kernel<<<lnGrid, blk, 0, stream>>>(h, ao, ln1g + l * D_MODEL,
                                              ln1b + l * D_MODEL, h);
        gemm_kernel<<<gF, blk, 0, stream>>>(h, W1_l, b1 + l * DFF, ff,
                                            MROWS, DFF, D_MODEL, 1);
        gemm_kernel<<<gD, blk, 0, stream>>>(ff, W2_l, b2 + l * D_MODEL, ao,
                                            MROWS, D_MODEL, DFF, 0);
        ln_kernel<<<lnGrid, blk, 0, stream>>>(h, ao, ln2g + l * D_MODEL,
                                              ln2b + l * D_MODEL, h);
    }

    head_kernel<<<BATCH, blk, 0, stream>>>(h, lng, lnb, fcw, fcb, out);
    (void)in_sizes; (void)n_in; (void)out_size; (void)ws_size;
}
